// LoRAAttention_33251636806060
// MI455X (gfx1250) — compile-verified, matrix operand order checked by reading, not run on hardware
//
#include <hip/hip_runtime.h>

// ---------- types ----------
typedef __bf16 bf16_t;
typedef __bf16 v16bf __attribute__((ext_vector_type(16)));
typedef float  v8f   __attribute__((ext_vector_type(8)));
typedef unsigned int v4u __attribute__((ext_vector_type(4)));
typedef unsigned int v8u __attribute__((ext_vector_type(8)));
typedef int g4i __attribute__((vector_size(16)));   // matches async-LDS builtin pointee

#define B_  16
#define N_  1024
#define C_  768
#define H_  12
#define HD_ 64
#define R_  16
#define SCALE_ 0.125f   // HD^-0.5

// ---- CDNA5 data-movement feature gates ----
#ifndef USE_TDM
#define USE_TDM 1       // tensor_load_to_lds for attention K tiles
#endif

#if defined(__has_builtin)
#if __has_builtin(__builtin_amdgcn_global_load_async_to_lds_b128)
#define USE_ASYNC_COPY 1
#endif
#endif
#ifndef USE_ASYNC_COPY
#define USE_ASYNC_COPY 0
#endif

__device__ __forceinline__ unsigned short f32_to_bf16(float f) {
  unsigned int u = __float_as_uint(f);
  unsigned int r = (u + 0x7FFFu + ((u >> 16) & 1u)) >> 16;
  return (unsigned short)r;
}

__device__ __forceinline__ v8f wmma_bf16(v16bf a, v16bf b, v8f c) {
  return __builtin_amdgcn_wmma_f32_16x16x32_bf16(false, a, false, b, (short)0, c, false, false);
}

#if USE_ASYNC_COPY
// 16B global -> LDS async copy per lane (ASYNCcnt tracked)
__device__ __forceinline__ void async16(const bf16_t* g, bf16_t* l) {
  __builtin_amdgcn_global_load_async_to_lds_b128(
      (__attribute__((address_space(1))) g4i*)g,
      (__attribute__((address_space(3))) g4i*)l, 0, 0);
}
__device__ __forceinline__ void wait_async0() {
  asm volatile("s_wait_asynccnt 0x0" ::: "memory");
}
#endif

#if USE_TDM
__device__ __forceinline__ unsigned lds_offset_of(const void* p) {
  return (unsigned)(unsigned long long)(__attribute__((address_space(3))) const void*)p;
}
#endif

// ---------- small prep kernels (fold LoRA into effective weights) ----------
__global__ void cast_x_kernel(const float* __restrict__ x, unsigned int* __restrict__ o) {
  int i = blockIdx.x * 256 + threadIdx.x;            // n2 threads, exact
  float a = x[2 * i], b = x[2 * i + 1];
  o[i] = (unsigned int)f32_to_bf16(a) | ((unsigned int)f32_to_bf16(b) << 16);
}

__global__ void prep_wqkv_kernel(const float* __restrict__ W,
                                 const float* __restrict__ Aq, const float* __restrict__ Bq,
                                 const float* __restrict__ Ak, const float* __restrict__ Bk,
                                 const float* __restrict__ Av, const float* __restrict__ Bv,
                                 unsigned short* __restrict__ outw) {
  int idx = blockIdx.x * 256 + threadIdx.x;          // 2304*768 threads, exact
  int i = idx / C_, j = idx % C_;
  int sel = i / C_, ir = i % C_;
  const float* Bp = (sel == 0) ? Bq : ((sel == 1) ? Bk : Bv);
  const float* Ap = (sel == 0) ? Aq : ((sel == 1) ? Ak : Av);
  float s = W[idx];
#pragma unroll
  for (int r = 0; r < R_; ++r) s += Bp[ir * R_ + r] * Ap[r * C_ + j];
  outw[idx] = f32_to_bf16(s);
}

__global__ void prep_T_kernel(const float* __restrict__ Ao, const float* __restrict__ Wp,
                              const float* __restrict__ bp,
                              float* __restrict__ T, float* __restrict__ tb) {
  int idx = blockIdx.x * 256 + threadIdx.x;          // 16*768 threads, exact
  int r = idx / C_, j = idx % C_;
  float s = 0.f;
  for (int c = 0; c < C_; ++c) s += Ao[r * C_ + c] * Wp[c * C_ + j];
  T[idx] = s;
  if (idx < R_) {
    float t = 0.f;
    for (int c = 0; c < C_; ++c) t += Ao[idx * C_ + c] * bp[c];
    tb[idx] = t;
  }
}

__global__ void prep_wproj_kernel(const float* __restrict__ Wp, const float* __restrict__ Bo,
                                  const float* __restrict__ T, const float* __restrict__ bp,
                                  const float* __restrict__ tb,
                                  unsigned short* __restrict__ outw, float* __restrict__ beff) {
  int idx = blockIdx.x * 256 + threadIdx.x;          // 768*768 threads, exact
  int i = idx / C_, j = idx % C_;
  float s = Wp[idx];
#pragma unroll
  for (int r = 0; r < R_; ++r) s += Bo[i * R_ + r] * T[r * C_ + j];
  outw[idx] = f32_to_bf16(s);
  if (j == 0) {
    float b = bp[i];
#pragma unroll
    for (int r = 0; r < R_; ++r) b += Bo[i * R_ + r] * tb[r];
    beff[i] = b;
  }
}

// ---------- tiled WMMA GEMM: D[M,Ntot] = A[M,K] @ Bt[Ntot,K]^T + bias ----------
// block tile 128x64, K-step 32, 256 threads = 8 waves (4x2), wave tile 32x32.
#define LDSA 48   // 32 K + 16 pad elems -> 96B row stride (32B aligned frags)
template <int MODE>
__global__ __launch_bounds__(256) void wmma_gemm_kernel(
    const bf16_t* __restrict__ A, const bf16_t* __restrict__ Bt,
    const float* __restrict__ bias,
    bf16_t* __restrict__ qb, bf16_t* __restrict__ kb, bf16_t* __restrict__ vb,
    float* __restrict__ outF, int K) {
  __shared__ alignas(32) bf16_t lA[128 * LDSA];
  __shared__ alignas(32) bf16_t lB[64 * LDSA];

  const int tid = threadIdx.x;
  const int rowBase = blockIdx.y * 128;
  const int nBase = blockIdx.x * 64;
  const int w = tid >> 5, lane = tid & 31;
  const int wm = w >> 1, wn = w & 1;
  const int lhalf = lane >> 4, l16 = lane & 15;

  const int arow = tid >> 1, ahalf = (tid & 1) * 16;     // A tile: 16 elems/thread
  const int brow = tid >> 2, bq = (tid & 3) * 8;         // B tile: 8 elems/thread

  v8f acc[2][2];
  const v8f z8 = {0.f, 0.f, 0.f, 0.f, 0.f, 0.f, 0.f, 0.f};
  acc[0][0] = z8; acc[0][1] = z8; acc[1][0] = z8; acc[1][1] = z8;

  for (int k0 = 0; k0 < K; k0 += 32) {
    __syncthreads();
    const bf16_t* gA = A + (size_t)(rowBase + arow) * K + k0 + ahalf;
    const bf16_t* gB = Bt + (size_t)(nBase + brow) * K + k0 + bq;
#if USE_ASYNC_COPY
    // async global->LDS staging: no VGPR round-trip, ASYNCcnt tracked
    async16(gA, &lA[arow * LDSA + ahalf]);
    async16(gA + 8, &lA[arow * LDSA + ahalf + 8]);
    async16(gB, &lB[brow * LDSA + bq]);
    wait_async0();
#else
    const uint4* ga = (const uint4*)gA;
    uint4 a0 = ga[0], a1 = ga[1];
    uint4 b0 = *(const uint4*)gB;
    *(uint4*)&lA[arow * LDSA + ahalf] = a0;
    *(uint4*)&lA[arow * LDSA + ahalf + 8] = a1;
    *(uint4*)&lB[brow * LDSA + bq] = b0;
#endif
    __syncthreads();

    if (k0 + 32 < K) {   // pull next K-tile toward the caches (global_prefetch_b8)
      __builtin_prefetch(gA + 32, 0, 1);
      __builtin_prefetch(gB + 32, 0, 1);
    }

    v16bf af0 = *(const v16bf*)&lA[(wm * 32 + l16) * LDSA + lhalf * 16];
    v16bf af1 = *(const v16bf*)&lA[(wm * 32 + 16 + l16) * LDSA + lhalf * 16];
    v16bf bf0 = *(const v16bf*)&lB[(wn * 32 + l16) * LDSA + lhalf * 16];
    v16bf bf1 = *(const v16bf*)&lB[(wn * 32 + 16 + l16) * LDSA + lhalf * 16];

    acc[0][0] = wmma_bf16(af0, bf0, acc[0][0]);
    acc[0][1] = wmma_bf16(af0, bf1, acc[0][1]);
    acc[1][0] = wmma_bf16(af1, bf0, acc[1][0]);
    acc[1][1] = wmma_bf16(af1, bf1, acc[1][1]);
  }

#pragma unroll
  for (int mi = 0; mi < 2; ++mi)
#pragma unroll
    for (int ni = 0; ni < 2; ++ni) {
      int col = nBase + wn * 32 + ni * 16 + l16;
      float bcol = bias[col];
#pragma unroll
      for (int v = 0; v < 8; ++v) {
        int row = rowBase + wm * 32 + mi * 16 + v + 8 * lhalf;
        float val = acc[mi][ni][v] + bcol;
        if (MODE == 0) {
          int sel = col / C_, jr = col % C_;
          int h = jr >> 6, d = jr & 63;
          int bb = row >> 10, t = row & 1023;
          bf16_t* dst = (sel == 0) ? qb : ((sel == 1) ? kb : vb);
          ((unsigned short*)dst)[((size_t)(bb * H_ + h) * N_ + t) * HD_ + d] = f32_to_bf16(val);
        } else {
          outF[(size_t)row * C_ + col] = val;
        }
      }
    }
}

// ---------- flash attention: one (b,h), 128 query rows per block ----------
#define LDSV 80   // 64 + 16 pad elems -> 160B row stride
__global__ __launch_bounds__(256) void flash_attn_kernel(
    const bf16_t* __restrict__ qb, const bf16_t* __restrict__ kb,
    const bf16_t* __restrict__ vb, bf16_t* __restrict__ attn) {
  __shared__ alignas(32) bf16_t lK[64 * LDSV];
  __shared__ alignas(32) bf16_t lV[64 * LDSV];            // transposed: [d][kv]
  __shared__ alignas(32) bf16_t lP[8 * 16 * LDSV];        // per-wave P staging

  const int tid = threadIdx.x, lane = tid & 31, w = tid >> 5;
  const int lhalf = lane >> 4, l16 = lane & 15;
  const int b = blockIdx.z, h = blockIdx.y, qBase = blockIdx.x * 128;
  const size_t headOff = (size_t)(b * H_ + h) * N_ * HD_;
  const bf16_t* qh = qb + headOff;
  const bf16_t* kh = kb + headOff;
  const bf16_t* vh = vb + headOff;

  // Q A-fragments, resident for the whole kernel
  v16bf aq0, aq1;
  {
    const bf16_t* qp = qh + (size_t)(qBase + w * 16 + l16) * HD_ + lhalf * 16;
    aq0 = *(const v16bf*)(qp);
    aq1 = *(const v16bf*)(qp + 32);
  }

  const v8f z8 = {0.f, 0.f, 0.f, 0.f, 0.f, 0.f, 0.f, 0.f};
  v8f accO[4];
  accO[0] = z8; accO[1] = z8; accO[2] = z8; accO[3] = z8;
  float mrow[8], lrow[8];
#pragma unroll
  for (int v = 0; v < 8; ++v) { mrow[v] = -3.0e38f; lrow[v] = 0.f; }

  const int krow = tid >> 2, kq = (tid & 3) * 16;         // 16 elems/thread per tile

#if USE_TDM
  const unsigned lk_base = lds_offset_of(lK);
#endif

  for (int kv0 = 0; kv0 < N_; kv0 += 64) {
    __syncthreads();
#if USE_TDM
    // K tile via Tensor Data Mover: 2D 64x64 bf16 tile, HW row padding
    // pad_interval=4 (every 32 DWORDs = one 128B row), pad_amount=7 (+8 DWORDs = 32B)
    // -> LDS row stride 160B == LDSV, matching the fragment addressing.
    {
      unsigned long long ga = (unsigned long long)(const void*)(kh + (size_t)kv0 * HD_);
      v4u g0;
      g0.x = 1u;                                            // count=1 (valid D#)
      g0.y = lk_base;                                       // lds_addr
      g0.z = (unsigned)ga;                                  // global_addr[31:0]
      g0.w = (unsigned)((ga >> 32) & 0x01FFFFFFull) | (2u << 30);  // addr[56:32], type=2
      v8u g1;
      g1[0] = (1u << 16) | (1u << 20) | (4u << 22) | (7u << 25);   // 2B elems, pad en
      g1[1] = (64u << 16);                                  // tensor_dim0 = 64
      g1[2] = ((unsigned)N_ << 16);                         // tensor_dim1 = 1024
      g1[3] = (64u << 16);                                  // tile_dim0 = 64
      g1[4] = 64u;                                          // tile_dim1 = 64
      g1[5] = 64u;                                          // tensor_dim0_stride = 64
      g1[6] = 0u;
      g1[7] = 0u;
      v4u gz = {0u, 0u, 0u, 0u};
      if (w == 0) {
        asm volatile("tensor_load_to_lds %0, %1, %2, %3"
                     :: "s"(g0), "s"(g1), "s"(gz), "s"(gz) : "memory");
        __builtin_amdgcn_s_wait_tensorcnt(0);
      }
    }
#else
    {
      const uint4* gk = (const uint4*)(kh + (size_t)(kv0 + krow) * HD_ + kq);
      uint4 k0 = gk[0], k1 = gk[1];
      *(uint4*)&lK[krow * LDSV + kq] = k0;
      *(uint4*)&lK[krow * LDSV + kq + 8] = k1;
    }
#endif
    // V tile, transposed into [d][kv] (manual: needs transpose on the way in)
    {
      const uint4* gv = (const uint4*)(vh + (size_t)(kv0 + krow) * HD_ + kq);
      uint4 v0 = gv[0], v1 = gv[1];
      const bf16_t* pv0 = (const bf16_t*)&v0;
      const bf16_t* pv1 = (const bf16_t*)&v1;
#pragma unroll
      for (int e = 0; e < 8; ++e) lV[(kq + e) * LDSV + krow] = pv0[e];
#pragma unroll
      for (int e = 0; e < 8; ++e) lV[(kq + 8 + e) * LDSV + krow] = pv1[e];
    }
    __syncthreads();

    // S = Q @ K^T (16 rows x 64 kv cols per wave)
    v8f accS[4];
#pragma unroll
    for (int c = 0; c < 4; ++c) {
      accS[c] = z8;
      v16bf bk0 = *(const v16bf*)&lK[(c * 16 + l16) * LDSV + lhalf * 16];
      v16bf bk1 = *(const v16bf*)&lK[(c * 16 + l16) * LDSV + 32 + lhalf * 16];
      accS[c] = wmma_bf16(aq0, bk0, accS[c]);
      accS[c] = wmma_bf16(aq1, bk1, accS[c]);
    }

    // online softmax (rows live in vgpr v + 8*lhalf; 16 lanes/half hold the cols)
#pragma unroll
    for (int c = 0; c < 4; ++c)
#pragma unroll
      for (int v = 0; v < 8; ++v) accS[c][v] *= SCALE_;

#pragma unroll
    for (int v = 0; v < 8; ++v) {
      float mx = fmaxf(fmaxf(accS[0][v], accS[1][v]), fmaxf(accS[2][v], accS[3][v]));
      for (int off = 8; off >= 1; off >>= 1) mx = fmaxf(mx, __shfl_xor(mx, off, 32));
      float mn = fmaxf(mrow[v], mx);
      float sc = __expf(mrow[v] - mn);
      mrow[v] = mn;
      float rs = 0.f;
#pragma unroll
      for (int c = 0; c < 4; ++c) { float p = __expf(accS[c][v] - mn); accS[c][v] = p; rs += p; }
      for (int off = 8; off >= 1; off >>= 1) rs += __shfl_xor(rs, off, 32);
      lrow[v] = lrow[v] * sc + rs;
#pragma unroll
      for (int o = 0; o < 4; ++o) accO[o][v] *= sc;
    }

    // stage P (C-layout -> LDS) then reload in A-layout
    bf16_t* lPw = lP + w * 16 * LDSV;
#pragma unroll
    for (int c = 0; c < 4; ++c)
#pragma unroll
      for (int v = 0; v < 8; ++v)
        ((unsigned short*)lPw)[(v + 8 * lhalf) * LDSV + c * 16 + l16] = f32_to_bf16(accS[c][v]);

#pragma unroll
    for (int kk = 0; kk < 2; ++kk) {
      v16bf ap = *(const v16bf*)&lPw[l16 * LDSV + kk * 32 + lhalf * 16];
#pragma unroll
      for (int o = 0; o < 4; ++o) {
        v16bf bv = *(const v16bf*)&lV[(o * 16 + l16) * LDSV + kk * 32 + lhalf * 16];
        accO[o] = wmma_bf16(ap, bv, accO[o]);
      }
    }
  }

  // finalize: O /= l, write bf16 to attn_out [B,N,C] at column h*64
#pragma unroll
  for (int v = 0; v < 8; ++v) {
    float inv = 1.0f / lrow[v];
    int t = qBase + w * 16 + v + 8 * lhalf;
#pragma unroll
    for (int o = 0; o < 4; ++o) {
      int d = o * 16 + l16;
      ((unsigned short*)attn)[(size_t)(b * N_ + t) * C_ + h * HD_ + d] =
          f32_to_bf16(accO[o][v] * inv);
    }
  }
}

// ---------- host ----------
extern "C" void kernel_launch(void* const* d_in, const int* in_sizes, int n_in,
                              void* d_out, int out_size, void* d_ws, size_t ws_size,
                              hipStream_t stream) {
  (void)in_sizes; (void)n_in; (void)out_size; (void)ws_size;
  const float* x      = (const float*)d_in[0];
  const float* W_qkv  = (const float*)d_in[1];
  const float* b_qkv  = (const float*)d_in[2];
  const float* W_proj = (const float*)d_in[3];
  const float* b_proj = (const float*)d_in[4];
  const float* A_q = (const float*)d_in[5];
  const float* B_q = (const float*)d_in[6];
  const float* A_k = (const float*)d_in[7];
  const float* B_k = (const float*)d_in[8];
  const float* A_v = (const float*)d_in[9];
  const float* B_v = (const float*)d_in[10];
  const float* A_o = (const float*)d_in[11];
  const float* B_o = (const float*)d_in[12];
  float* out = (float*)d_out;

  // workspace carve-up (all offsets 16B-aligned; bf16 GEMM bases 32B-aligned)
  char* ws = (char*)d_ws;
  const size_t TOK = (size_t)B_ * N_;              // 16384
  bf16_t* x_bf  = (bf16_t*)ws;                                   // TOK*C
  bf16_t* wqkv  = x_bf + TOK * C_;                               // 3C*C
  bf16_t* wproj = wqkv + (size_t)3 * C_ * C_;                    // C*C
  float*  beff  = (float*)(wproj + (size_t)C_ * C_);             // C
  float*  T     = beff + C_;                                     // R*C
  float*  tb    = T + (size_t)R_ * C_;                           // 16
  bf16_t* qbuf  = (bf16_t*)(tb + 16);                            // TOK*C
  bf16_t* kbuf  = qbuf + TOK * C_;
  bf16_t* vbuf  = kbuf + TOK * C_;
  bf16_t* attn  = vbuf + TOK * C_;

  // 1. cast x -> bf16
  cast_x_kernel<<<(TOK * C_ / 2) / 256, 256, 0, stream>>>(x, (unsigned int*)x_bf);
  // 2. fold LoRA into effective weights
  prep_wqkv_kernel<<<(3 * C_ * C_) / 256, 256, 0, stream>>>(
      W_qkv, A_q, B_q, A_k, B_k, A_v, B_v, (unsigned short*)wqkv);
  prep_T_kernel<<<(R_ * C_) / 256, 256, 0, stream>>>(A_o, W_proj, b_proj, T, tb);
  prep_wproj_kernel<<<(C_ * C_) / 256, 256, 0, stream>>>(
      W_proj, B_o, T, b_proj, tb, (unsigned short*)wproj, beff);
  // 3. fused QKV GEMM (LoRA folded), scatter to head layout
  wmma_gemm_kernel<0><<<dim3((3 * C_) / 64, TOK / 128), 256, 0, stream>>>(
      x_bf, wqkv, b_qkv, qbuf, kbuf, vbuf, nullptr, C_);
  // 4. flash attention
  flash_attn_kernel<<<dim3(N_ / 128, H_, B_), 256, 0, stream>>>(qbuf, kbuf, vbuf, attn);
  // 5. projection GEMM (output LoRA folded) -> fp32 out
  wmma_gemm_kernel<1><<<dim3(C_ / 64, TOK / 128), 256, 0, stream>>>(
      attn, wproj, beff, nullptr, nullptr, nullptr, out, C_);
}
